// Quantization_63763084477352
// MI455X (gfx1250) — matrix operand order repaired and checked by reading, not compiled, not run on hardware
//
#include <hip/hip_runtime.h>

typedef __attribute__((ext_vector_type(16))) __bf16   bf16x16;
typedef __attribute__((ext_vector_type(8)))  float    f32x8;
typedef __attribute__((ext_vector_type(4)))  unsigned int u32x4;

static constexpr int   N_PTS  = 131072;
static constexpr int   DIM    = 256;
static constexpr int   K_CB   = 1024;
static constexpr int   ROWS   = 128;          // z rows per workgroup
static constexpr int   ROWP   = DIM + 8;      // padded LDS row stride (bf16 elems) -> conflict-free
static constexpr float BETA   = 0.25f;

__device__ __forceinline__ unsigned short f2bf(float f) {
  unsigned int u = __float_as_uint(f);
  unsigned int r = u + 0x7FFFu + ((u >> 16) & 1u);   // round-to-nearest-even
  return (unsigned short)(r >> 16);
}
__device__ __forceinline__ float bf2f(unsigned short h) {
  return __uint_as_float(((unsigned int)h) << 16);
}
__device__ __forceinline__ bf16x16 make_frag(u32x4 a, u32x4 b) {
  union { bf16x16 v; u32x4 u[2]; } f;
  f.u[0] = a; f.u[1] = b;
  return f.v;
}

// ---------------- Kernel 1: codebook -> bf16 hi/lo + ||c||^2 ----------------
__global__ void __launch_bounds__(256)
vq_prep(const float* __restrict__ cb, unsigned short* __restrict__ cb_hi,
        unsigned short* __restrict__ cb_lo, float* __restrict__ cnorm) {
  __shared__ float red[8];
  const int k = blockIdx.x, tid = threadIdx.x;           // blockDim == DIM == 256
  float c = cb[k * DIM + tid];
  unsigned short h = f2bf(c);
  cb_hi[k * DIM + tid] = h;
  cb_lo[k * DIM + tid] = f2bf(c - bf2f(h));
  float s = c * c;
  #pragma unroll
  for (int off = 16; off >= 1; off >>= 1) s += __shfl_xor(s, off, 32);
  if ((tid & 31) == 0) red[tid >> 5] = s;
  __syncthreads();
  if (tid == 0) {
    float t = 0.f;
    #pragma unroll
    for (int w = 0; w < 8; ++w) t += red[w];
    cnorm[k] = t;
  }
}

// ---------------- Kernel 2: distances (WMMA) + argmin + gather + partial loss
__global__ void __launch_bounds__(256)
vq_main(const float* __restrict__ z, const float* __restrict__ cb,
        const unsigned short* __restrict__ cb_hi,
        const unsigned short* __restrict__ cb_lo,
        const float* __restrict__ cnorm,
        float* __restrict__ out_z, float* __restrict__ out_zq,
        float* __restrict__ out_ids, float* __restrict__ partial) {
  extern __shared__ unsigned char smem[];
  unsigned short* zhi = (unsigned short*)smem;                  // ROWS x ROWP
  unsigned short* zlo = zhi + ROWS * ROWP;                      // ROWS x ROWP
  int*   ids_tile = (int*)(zlo + ROWS * ROWP);                  // ROWS
  float* red      = (float*)(ids_tile + ROWS);                  // 8

  const int tid  = threadIdx.x;
  const int wave = tid >> 5;
  const int lane = tid & 31;
  const int half = lane >> 4;
  const int ln   = lane & 15;
  const int row0 = blockIdx.x * ROWS;

  // Stage z tile into LDS as bf16 hi/lo; emit passthrough output z.
  for (int i = tid; i < ROWS * DIM; i += 256) {
    int r = i >> 8, d = i & (DIM - 1);
    long gi = (long)(row0 + r) * DIM + d;
    float v = z[gi];
    out_z[gi] = v;
    unsigned short h = f2bf(v);
    zhi[r * ROWP + d] = h;
    zlo[r * ROWP + d] = f2bf(v - bf2f(h));
  }
  __syncthreads();

  float best[8];
  int   bid[8];
  #pragma unroll
  for (int r = 0; r < 8; ++r) { best[r] = 3.0e38f; bid[r] = 0; }

  // A-fragment base for this lane (A layout: lanes 0..15 hold M=0..15).
  const unsigned short* zh_row = zhi + (wave * 16 + ln) * ROWP;
  const unsigned short* zl_row = zlo + (wave * 16 + ln) * ROWP;

  for (int kc = 0; kc < K_CB / 16; ++kc) {
    const int code = kc * 16 + ln;                       // B layout: n = lane&15
    const unsigned short* bh = cb_hi + code * DIM + half * 16;
    const unsigned short* bl = cb_lo + code * DIM + half * 16;
    if (kc + 1 < K_CB / 16) {                            // next-chunk prefetch
      __builtin_prefetch(cb_hi + (code + 16) * DIM, 0, 0);
      __builtin_prefetch(cb_lo + (code + 16) * DIM, 0, 0);
    }
    f32x8 acc = {0.f, 0.f, 0.f, 0.f, 0.f, 0.f, 0.f, 0.f};
    #pragma unroll
    for (int s = 0; s < DIM / 32; ++s) {
      const int a0 = s * 32 + half * 16;                 // A: k = s*32 + half*16 + j
      bf16x16 a_hi = make_frag(*(const u32x4*)(zh_row + a0),
                               *(const u32x4*)(zh_row + a0 + 8));
      bf16x16 a_lo = make_frag(*(const u32x4*)(zl_row + a0),
                               *(const u32x4*)(zl_row + a0 + 8));
      bf16x16 b_hi = *(const bf16x16*)(bh + s * 32);     // B: k = s*32 + half*16 + j
      bf16x16 b_lo = *(const bf16x16*)(bl + s * 32);
      acc = __builtin_amdgcn_wmma_f32_16x16x32_bf16(false, a_hi, false, b_hi,
                                                    (short)0, acc, false, false);
      acc = __builtin_amdgcn_wmma_f32_16x16x32_bf16(false, a_hi, false, b_lo,
                                                    (short)0, acc, false, false);
      acc = __builtin_amdgcn_wmma_f32_16x16x32_bf16(false, a_lo, false, b_hi,
                                                    (short)0, acc, false, false);
    }
    // score = ||c||^2 - 2 z.c   (||z||^2 constant per row -> irrelevant to argmin)
    const float cn = cnorm[code];
    #pragma unroll
    for (int r = 0; r < 8; ++r) {
      float sc = __builtin_fmaf(-2.0f, acc[r], cn);
      if (sc < best[r]) { best[r] = sc; bid[r] = code; }
    }
  }

  // Cross-lane argmin within each 16-lane half group (D row m = half*8 + r).
  #pragma unroll
  for (int r = 0; r < 8; ++r) {
    float v = best[r];
    int   id = bid[r];
    #pragma unroll
    for (int off = 8; off >= 1; off >>= 1) {
      float ov = __shfl_xor(v, off, 32);
      int   oi = __shfl_xor(id, off, 32);
      if (ov < v || (ov == v && oi < id)) { v = ov; id = oi; }
    }
    if (ln == 0) {
      int m = wave * 16 + half * 8 + r;
      ids_tile[m] = id;
      out_ids[row0 + m] = (float)id;                     // tuple outputs land as f32
    }
  }
  __syncthreads();

  // z_q gather from fp32 codebook + per-WG loss partial (sum of (zq - z)^2).
  float lsum = 0.f;
  for (int i = tid; i < ROWS * DIM; i += 256) {
    int r = i >> 8, d = i & (DIM - 1);
    long gi = (long)(row0 + r) * DIM + d;
    float q = cb[(long)ids_tile[r] * DIM + d];
    out_zq[gi] = q;
    float diff = q - z[gi];
    lsum = __builtin_fmaf(diff, diff, lsum);
  }
  #pragma unroll
  for (int off = 16; off >= 1; off >>= 1) lsum += __shfl_xor(lsum, off, 32);
  if (lane == 0) red[wave] = lsum;
  __syncthreads();
  if (tid == 0) {
    float t = 0.f;
    #pragma unroll
    for (int w = 0; w < 8; ++w) t += red[w];
    partial[blockIdx.x] = t;
  }
}

// ---------------- Kernel 3: deterministic final loss reduce ----------------
__global__ void __launch_bounds__(256)
vq_loss(const float* __restrict__ partial, float* __restrict__ out_loss) {
  __shared__ float red[8];
  float s = 0.f;
  for (int i = threadIdx.x; i < N_PTS / ROWS; i += 256) s += partial[i];
  #pragma unroll
  for (int off = 16; off >= 1; off >>= 1) s += __shfl_xor(s, off, 32);
  if ((threadIdx.x & 31) == 0) red[threadIdx.x >> 5] = s;
  __syncthreads();
  if (threadIdx.x == 0) {
    float t = 0.f;
    #pragma unroll
    for (int w = 0; w < 8; ++w) t += red[w];
    out_loss[0] = t * (1.0f + BETA) / (float)((long)N_PTS * DIM);
  }
}

extern "C" void kernel_launch(void* const* d_in, const int* in_sizes, int n_in,
                              void* d_out, int out_size, void* d_ws, size_t ws_size,
                              hipStream_t stream) {
  (void)in_sizes; (void)n_in; (void)out_size; (void)ws_size;
  const float* z  = (const float*)d_in[0];
  const float* cb = (const float*)d_in[1];

  float* out      = (float*)d_out;
  float* out_z    = out;
  float* out_zq   = out + (long)N_PTS * DIM;
  float* out_ids  = out + 2L * N_PTS * DIM;
  float* out_loss = out + 2L * N_PTS * DIM + N_PTS;

  unsigned char*  ws     = (unsigned char*)d_ws;
  unsigned short* cb_hi  = (unsigned short*)ws;                              // 512 KB
  unsigned short* cb_lo  = (unsigned short*)(ws + (size_t)K_CB * DIM * 2);   // 512 KB
  float*          cnorm  = (float*)(ws + (size_t)K_CB * DIM * 4);            // 4 KB
  float*          part   = (float*)(ws + (size_t)K_CB * DIM * 4 + K_CB * 4); // 4 KB

  vq_prep<<<K_CB, 256, 0, stream>>>(cb, cb_hi, cb_lo, cnorm);

  size_t smem = (size_t)ROWS * ROWP * 2 * 2   // zhi + zlo (bf16)
              + (size_t)ROWS * 4              // ids_tile
              + 8 * 4;                        // red
  vq_main<<<N_PTS / ROWS, 256, smem, stream>>>(z, cb, cb_hi, cb_lo, cnorm,
                                               out_z, out_zq, out_ids, part);

  vq_loss<<<1, 256, 0, stream>>>(part, out_loss);
}
